// AttnDecoderRNN_24094766531225
// MI455X (gfx1250) — matrix operand order repaired and checked by reading, not compiled, not run on hardware
//
#include <hip/hip_runtime.h>
#include <hip/hip_bf16.h>
#include <math.h>

#define N_BATCH 64
#define T_ENC   800
#define H_DIM   512
#define H2_DIM  1024
#define O_DIM   80
#define O_PAD   96

typedef __attribute__((ext_vector_type(16))) _Float16 v16h;
typedef __attribute__((ext_vector_type(8)))  float    v8f;

__device__ __forceinline__ float sigmoidf_(float x) { return 1.0f / (1.0f + __expf(-x)); }

// ---------------------------------------------------------------------------
// C[n,m] = act( sum_k A[n,k] * W[m,k] + bias[m] )
// A: [64, K] row-major, W: [M, K] row-major. K must be a multiple of 32.
// block = 128 (4 waves); wave w owns column tile (blockIdx.x*4 + w)*16.
// grid = (ceil(M/64), 4). One 16x16 tile per wave via v_wmma_f32_16x16x32_f16.
// ---------------------------------------------------------------------------
__global__ void gemm_wmma_nt(const float* __restrict__ A,
                             const float* __restrict__ W,
                             const float* __restrict__ bias,
                             float* __restrict__ C,
                             int K, int M, int act)
{
    const int lane = threadIdx.x & 31;
    const int wave = threadIdx.x >> 5;
    const int g    = lane >> 4;   // half-wave group (0/1)
    const int r    = lane & 15;
    const int row0 = blockIdx.y * 16;                   // batch-row tile
    const int col0 = (blockIdx.x * 4 + wave) * 16;      // output-col tile
    if (col0 >= M) return;

    const float* arow = A + (size_t)(row0 + r) * K;     // batch row (A operand)
    const float* wrow = W + (size_t)(col0 + r) * K;     // weight row (B operand)

    v8f acc = {};
    for (int kt = 0; kt < K; kt += 32) {
        // A 16x32 f16 layout: lane(g,r): halves 0..7  -> K = kt+g*8+0..7
        //                               halves 8..15 -> K = kt+16+g*8+0..7
        const float4 a0 = *(const float4*)(arow + kt + g * 8);
        const float4 a1 = *(const float4*)(arow + kt + g * 8 + 4);
        const float4 a2 = *(const float4*)(arow + kt + 16 + g * 8);
        const float4 a3 = *(const float4*)(arow + kt + 16 + g * 8 + 4);
        // B 32x16 f16 layout: lane(g,r) holds column n=r, halves 0..7 -> K = kt+g*8+0..7, halves 8..15 -> K = kt+16+g*8+0..7
        const float4 b0 = *(const float4*)(wrow + kt + g * 8);
        const float4 b1 = *(const float4*)(wrow + kt + g * 8 + 4);
        const float4 b2 = *(const float4*)(wrow + kt + 16 + g * 8);
        const float4 b3 = *(const float4*)(wrow + kt + 16 + g * 8 + 4);

        v16h a, b;
        a[0]  = (_Float16)a0.x; a[1]  = (_Float16)a0.y; a[2]  = (_Float16)a0.z; a[3]  = (_Float16)a0.w;
        a[4]  = (_Float16)a1.x; a[5]  = (_Float16)a1.y; a[6]  = (_Float16)a1.z; a[7]  = (_Float16)a1.w;
        a[8]  = (_Float16)a2.x; a[9]  = (_Float16)a2.y; a[10] = (_Float16)a2.z; a[11] = (_Float16)a2.w;
        a[12] = (_Float16)a3.x; a[13] = (_Float16)a3.y; a[14] = (_Float16)a3.z; a[15] = (_Float16)a3.w;
        b[0]  = (_Float16)b0.x; b[1]  = (_Float16)b0.y; b[2]  = (_Float16)b0.z; b[3]  = (_Float16)b0.w;
        b[4]  = (_Float16)b1.x; b[5]  = (_Float16)b1.y; b[6]  = (_Float16)b1.z; b[7]  = (_Float16)b1.w;
        b[8]  = (_Float16)b2.x; b[9]  = (_Float16)b2.y; b[10] = (_Float16)b2.z; b[11] = (_Float16)b2.w;
        b[12] = (_Float16)b3.x; b[13] = (_Float16)b3.y; b[14] = (_Float16)b3.z; b[15] = (_Float16)b3.w;

        acc = __builtin_amdgcn_wmma_f32_16x16x32_f16(
                  false, a, false, b, (short)0, acc, false, false);
    }

    const float bv = bias ? bias[col0 + r] : 0.0f;
    // D layout: lane(g,r), VGPR v -> D[m = v + 8g, n = r]
    #pragma unroll
    for (int v = 0; v < 8; ++v) {
        int row = row0 + v + 8 * g;
        float val = acc[v] + bv;
        if (act == 1) val = fmaxf(val, 0.0f);
        C[(size_t)row * M + col0 + r] = val;
    }
}

// Zero-pad the K dimension: dst[r, 0..Kp) = k<K ? src[r,k] : 0  (rows*Kp threads)
__global__ void pad_rows(const float* __restrict__ src, float* __restrict__ dst,
                         int K, int Kp, int total)
{
    int idx = blockIdx.x * 256 + threadIdx.x;
    if (idx >= total) return;
    int rrow = idx / Kp, k = idx - rrow * Kp;
    dst[idx] = (k < K) ? src[(size_t)rrow * K + k] : 0.0f;
}

// ---------------------------------------------------------------------------
// GRU cell pointwise: h' = (1-z)*n + z*h, gates (r,z,n). 64*1024 threads.
// ---------------------------------------------------------------------------
__global__ void gru_pointwise(const float* __restrict__ gi,
                              const float* __restrict__ gh,
                              const float* __restrict__ h,
                              float* __restrict__ hnew)
{
    int idx = blockIdx.x * blockDim.x + threadIdx.x;   // n*H2 + j
    int n = idx >> 10, j = idx & (H2_DIM - 1);
    const float* gin = gi + (size_t)n * 3 * H2_DIM;
    const float* ghn = gh + (size_t)n * 3 * H2_DIM;
    float rg = sigmoidf_(gin[j] + ghn[j]);
    float zg = sigmoidf_(gin[H2_DIM + j] + ghn[H2_DIM + j]);
    float ng = tanhf(gin[2 * H2_DIM + j] + rg * ghn[2 * H2_DIM + j]);
    hnew[idx] = (1.0f - zg) * ng + zg * h[idx];
}

// ---------------------------------------------------------------------------
// Attention scores: score[n,t] = t<len ? exp(dot(tanh(aw[n,t,:]+dp[n,:]),Wa)+ba) : 0
// block = 256 (8 waves), each wave one t. grid = (T/8, N). NT loads stream aw.
// ---------------------------------------------------------------------------
__global__ void attn_score(const float* __restrict__ aw,
                           const float* __restrict__ dp,
                           const float* __restrict__ Wattn,
                           const float* __restrict__ battn,
                           const int*   __restrict__ len,
                           float* __restrict__ score)
{
    int n    = blockIdx.y;
    int wave = threadIdx.x >> 5;
    int lane = threadIdx.x & 31;
    int t    = blockIdx.x * 8 + wave;
    int L    = len[n];
    if (t >= L) { if (lane == 0) score[n * T_ENC + t] = 0.0f; return; }

    const float* row = aw + ((size_t)n * T_ENC + t) * H2_DIM;
    const float* dpr = dp + (size_t)n * H2_DIM;
    float acc = 0.0f;
    #pragma unroll 4
    for (int i = 0; i < 32; ++i) {
        int hh = lane + i * 32;                 // coalesced across the wave
        float e = tanhf(__builtin_nontemporal_load(&row[hh]) + dpr[hh]);
        acc += e * Wattn[hh];
    }
    #pragma unroll
    for (int m = 16; m > 0; m >>= 1) acc += __shfl_xor(acc, m, 32);
    if (lane == 0) score[n * T_ENC + t] = __expf(acc + battn[0]);
}

// One block per batch row: denom[n] = max(sum_t score, 1e-12). No float atomics.
__global__ void attn_denom(const float* __restrict__ score, float* __restrict__ denom)
{
    __shared__ float red[256];
    int n = blockIdx.x;
    float acc = 0.0f;
    for (int t = threadIdx.x; t < T_ENC; t += 256) acc += score[n * T_ENC + t];
    red[threadIdx.x] = acc;
    __syncthreads();
    for (int s = 128; s > 0; s >>= 1) {
        if (threadIdx.x < s) red[threadIdx.x] += red[threadIdx.x + s];
        __syncthreads();
    }
    if (threadIdx.x == 0) denom[n] = fmaxf(red[0], 1e-12f);
}

// dec_in[n, h] = (sum_{t<len} score[n,t] * enc[n,t,h]) / denom[n]
// grid = (H2/256, N), block = 256. Streams enc with NT loads.
__global__ void attn_apply(const float* __restrict__ enc,
                           const float* __restrict__ score,
                           const float* __restrict__ denom,
                           const int*   __restrict__ len,
                           float* __restrict__ dec_in)
{
    int n = blockIdx.y;
    int h = blockIdx.x * 256 + threadIdx.x;
    int L = len[n];
    const float* srow = score + (size_t)n * T_ENC;
    const float* erow = enc + (size_t)n * T_ENC * H2_DIM + h;
    float acc = 0.0f;
    #pragma unroll 4
    for (int t = 0; t < L; ++t)
        acc += srow[t] * __builtin_nontemporal_load(&erow[(size_t)t * H2_DIM]);
    dec_in[(size_t)n * (2 * H2_DIM) + h] = acc / denom[n];
}

// dec_in[n, H2 + j] = h_att[n, j]
__global__ void concat_h(const float* __restrict__ h, float* __restrict__ dec_in)
{
    int idx = blockIdx.x * blockDim.x + threadIdx.x;   // 64*1024
    int n = idx >> 10, j = idx & (H2_DIM - 1);
    dec_in[(size_t)n * (2 * H2_DIM) + H2_DIM + j] = h[idx];
}

// b += a  (64*1024 elements)
__global__ void add_into(const float* __restrict__ a, float* __restrict__ b)
{
    int idx = blockIdx.x * blockDim.x + threadIdx.x;
    b[idx] += a[idx];
}

// ---------------------------------------------------------------------------
extern "C" void kernel_launch(void* const* d_in, const int* in_sizes, int n_in,
                              void* d_out, int out_size, void* d_ws, size_t ws_size,
                              hipStream_t stream)
{
    const float* input_enc      = (const float*)d_in[0];
    const float* input_attW_enc = (const float*)d_in[1];
    const float* input_dec      = (const float*)d_in[2];
    const int*   lengths_enc    = (const int*)  d_in[3];
    const float* hidden_att     = (const float*)d_in[4];
    const float* hidden_dec1    = (const float*)d_in[5];
    const float* hidden_dec2    = (const float*)d_in[6];
    const float* W_pre1 = (const float*)d_in[7];  const float* b_pre1 = (const float*)d_in[8];
    const float* W_pre2 = (const float*)d_in[9];  const float* b_pre2 = (const float*)d_in[10];
    const float* Wih_att = (const float*)d_in[11]; const float* Whh_att = (const float*)d_in[12];
    const float* bih_att = (const float*)d_in[13]; const float* bhh_att = (const float*)d_in[14];
    const float* W_ld  = (const float*)d_in[15]; const float* b_ld  = (const float*)d_in[16];
    const float* W_attn = (const float*)d_in[17]; const float* b_attn = (const float*)d_in[18];
    const float* W_sc  = (const float*)d_in[19]; const float* b_sc  = (const float*)d_in[20];
    const float* Wih_d1 = (const float*)d_in[21]; const float* Whh_d1 = (const float*)d_in[22];
    const float* bih_d1 = (const float*)d_in[23]; const float* bhh_d1 = (const float*)d_in[24];
    const float* Wih_d2 = (const float*)d_in[25]; const float* Whh_d2 = (const float*)d_in[26];
    const float* bih_d2 = (const float*)d_in[27]; const float* bhh_d2 = (const float*)d_in[28];
    const float* W_out = (const float*)d_in[29]; const float* b_out = (const float*)d_in[30];

    float* out = (float*)d_out;
    float* out_y  = out;                         // [64,2,80] = 10240
    float* h_att  = out + 10240;                 // [64,1024]
    float* h_dec1 = out + 10240 + 65536;         // [64,1024]
    float* h_dec2 = out + 10240 + 2 * 65536;     // [64,1024]

    float* ws = (float*)d_ws;
    float* pre1     = ws;                 // 64*1024
    float* pre2     = ws + 65536;         // 64*512
    float* gi       = ws + 98304;         // 64*3072
    float* gh       = ws + 294912;        // 64*3072
    float* dec_proj = ws + 491520;        // 64*1024
    float* score    = ws + 557056;        // 64*800
    float* denom    = ws + 608256;        // 64
    float* dec_in   = ws + 608320;        // 64*2048
    float* resid    = ws + 739392;        // 64*1024
    float* a_pad    = ws + 804928;        // 64*96
    float* w_pad    = ws + 811072;        // 1024*96

    const dim3 blk128(128), blk256(256);
    const dim3 grid_ew(64 * H2_DIM / 256);
    #define GEMM_GRID(M) dim3(((M) + 63) / 64, 4)

    // --- prenet (pad K: 80 -> 96 so the GEMM has no tail) ---
    pad_rows<<<dim3((N_BATCH * O_PAD + 255) / 256), blk256, 0, stream>>>(input_dec, a_pad, O_DIM, O_PAD, N_BATCH * O_PAD);
    pad_rows<<<dim3((H2_DIM * O_PAD + 255) / 256), blk256, 0, stream>>>(W_pre1, w_pad, O_DIM, O_PAD, H2_DIM * O_PAD);
    gemm_wmma_nt<<<GEMM_GRID(H2_DIM), blk128, 0, stream>>>(a_pad, w_pad, b_pre1, pre1, O_PAD, H2_DIM, 1);
    gemm_wmma_nt<<<GEMM_GRID(H_DIM),  blk128, 0, stream>>>(pre1, W_pre2, b_pre2, pre2, H2_DIM, H_DIM, 1);

    // --- attention GRU cell ---
    gemm_wmma_nt<<<GEMM_GRID(3 * H2_DIM), blk128, 0, stream>>>(pre2, Wih_att, bih_att, gi, H_DIM, 3 * H2_DIM, 0);
    gemm_wmma_nt<<<GEMM_GRID(3 * H2_DIM), blk128, 0, stream>>>(hidden_att, Whh_att, bhh_att, gh, H2_DIM, 3 * H2_DIM, 0);
    gru_pointwise<<<grid_ew, blk256, 0, stream>>>(gi, gh, hidden_att, h_att);

    // --- additive attention ---
    gemm_wmma_nt<<<GEMM_GRID(H2_DIM), blk128, 0, stream>>>(h_att, W_ld, b_ld, dec_proj, H2_DIM, H2_DIM, 0);
    attn_score<<<dim3(T_ENC / 8, N_BATCH), blk256, 0, stream>>>(input_attW_enc, dec_proj, W_attn, b_attn, lengths_enc, score);
    attn_denom<<<dim3(N_BATCH), blk256, 0, stream>>>(score, denom);
    attn_apply<<<dim3(H2_DIM / 256, N_BATCH), blk256, 0, stream>>>(input_enc, score, denom, lengths_enc, dec_in);
    concat_h<<<grid_ew, blk256, 0, stream>>>(h_att, dec_in);

    // --- short-cut + decoder GRU 1 ---
    gemm_wmma_nt<<<GEMM_GRID(H2_DIM), blk128, 0, stream>>>(dec_in, W_sc, b_sc, resid, 2 * H2_DIM, H2_DIM, 0);
    gemm_wmma_nt<<<GEMM_GRID(3 * H2_DIM), blk128, 0, stream>>>(dec_in, Wih_d1, bih_d1, gi, 2 * H2_DIM, 3 * H2_DIM, 0);
    gemm_wmma_nt<<<GEMM_GRID(3 * H2_DIM), blk128, 0, stream>>>(hidden_dec1, Whh_d1, bhh_d1, gh, H2_DIM, 3 * H2_DIM, 0);
    gru_pointwise<<<grid_ew, blk256, 0, stream>>>(gi, gh, hidden_dec1, h_dec1);
    add_into<<<grid_ew, blk256, 0, stream>>>(h_dec1, resid);

    // --- decoder GRU 2 ---
    gemm_wmma_nt<<<GEMM_GRID(3 * H2_DIM), blk128, 0, stream>>>(resid, Wih_d2, bih_d2, gi, H2_DIM, 3 * H2_DIM, 0);
    gemm_wmma_nt<<<GEMM_GRID(3 * H2_DIM), blk128, 0, stream>>>(hidden_dec2, Whh_d2, bhh_d2, gh, H2_DIM, 3 * H2_DIM, 0);
    gru_pointwise<<<grid_ew, blk256, 0, stream>>>(gi, gh, hidden_dec2, h_dec2);
    add_into<<<grid_ew, blk256, 0, stream>>>(h_dec2, resid);

    // --- output projection ---
    gemm_wmma_nt<<<GEMM_GRID(160), blk128, 0, stream>>>(resid, W_out, b_out, out_y, H2_DIM, 160, 0);
    #undef GEMM_GRID
}
